// SoftSuperpixelNeighborhoodAttention_27307402068262
// MI455X (gfx1250) — matrix operand order repaired and checked by reading, not compiled, not run on hardware
//
#include <hip/hip_runtime.h>

typedef __attribute__((ext_vector_type(16))) __bf16 v16bf;
typedef __attribute__((ext_vector_type(8)))  float  v8f;
typedef __attribute__((ext_vector_type(4)))  float  f4;
typedef __attribute__((ext_vector_type(2)))  float  f2;
typedef __attribute__((ext_vector_type(4)))  int    i4v;
typedef __attribute__((ext_vector_type(2)))  int    i2v;

#define Bn 4
#define Cn 256
#define Hn 128
#define Wn 128
#define HWn (Hn*Wn)
#define NPIX (Bn*Hn*Wn)      /* 65536 */
#define HEADSn 8
#define HDn 32
#define KSn 9
#define SCALEF 0.17677669529663687f   /* 32^-0.5 */

// ---- CDNA5 async global->LDS copies (guarded; fallback = vector copies) ----
#if defined(__gfx1250__)
#if __has_builtin(__builtin_amdgcn_global_load_async_to_lds_b128) && \
    __has_builtin(__builtin_amdgcn_global_load_async_to_lds_b64)
#define ASYNC_LDS 1
#endif
#endif

#ifdef ASYNC_LDS
typedef __attribute__((address_space(1))) i4v* g128_t;
typedef __attribute__((address_space(3))) i4v* l128_t;
typedef __attribute__((address_space(1))) i2v* g64_t;
typedef __attribute__((address_space(3))) i2v* l64_t;
__device__ __forceinline__ void async_cp128(const void* g, void* l) {
  __builtin_amdgcn_global_load_async_to_lds_b128((g128_t)g, (l128_t)l, 0, 0);
}
__device__ __forceinline__ void async_cp64(const void* g, void* l) {
  __builtin_amdgcn_global_load_async_to_lds_b64((g64_t)g, (l64_t)l, 0, 0);
}
__device__ __forceinline__ void async_wait0() {
#if __has_builtin(__builtin_amdgcn_s_wait_asynccnt)
  __builtin_amdgcn_s_wait_asynccnt(0);
#else
  asm volatile("s_wait_asynccnt 0" ::: "memory");
#endif
}
#endif

__device__ __forceinline__ v8f wmma_bf16(v16bf a, v16bf b, v8f c) {
  // v_wmma_f32_16x16x32_bf16: D = A(16x32) * B(32x16) + C(16x16 f32)
  return __builtin_amdgcn_wmma_f32_16x16x32_bf16(false, a, false, b, (short)0, c,
                                                 false, false);
}

// A-fragment (16x32 bf16, row-major source, 64B rows): lane L holds row (L&15);
// lanes 0-15 take K 0..15 contiguous, lanes 16-31 take K 16..31.
__device__ __forceinline__ v16bf frag_ilv(const __bf16* row, int hh) {
  const f4* p = (const f4*)row;
  union { f4 f[2]; v16bf v; } u;
  u.f[0] = p[2 * hh];       // k = hh*16 .. hh*16+7
  u.f[1] = p[2 * hh + 1];   // k = hh*16+8 .. hh*16+15
  return u.v;
}

// B-fragment (32x16 bf16): lane L is column (L&15); lanes 0-15 hold K=0..15
// contiguous, lanes 16-31 hold K=16..31. Caller passes base + half*16 elements.
__device__ __forceinline__ v16bf frag_ctg(const __bf16* p16) {
  const f4* p = (const f4*)p16;
  union { f4 f[2]; v16bf v; } u;
  u.f[0] = p[0];
  u.f[1] = p[1];
  return u.v;
}

__device__ __forceinline__ unsigned short bfbits(float f) {
  __bf16 h = (__bf16)f;
  return __builtin_bit_cast(unsigned short, h);
}

// Stage one weight column (32 channels, compile-time stride -> immediate
// offsets in the global loads) into an LDS [n][c] row.
template <int STRIDE>
__device__ __forceinline__ void stage_wcol(const float* __restrict__ p,
                                           __bf16* __restrict__ dst) {
#pragma unroll
  for (int ci = 0; ci < 32; ++ci)
    dst[ci] = (__bf16)p[ci * STRIDE];
}

// ---------------------------------------------------------------------------
// Kernel 1: fused QKV projection.  A = x (BCHW f32) viewed as [pix][C],
// Wt = concat(w_qk, w_v) -> [pix][768].  Writes Q*scale, K as BHWC bf16 and
// V as BCHW bf16.  Block: 256 thr (8 waves). Tile M=64 x N=384; grid 1024x2.
// ---------------------------------------------------------------------------
__global__ __launch_bounds__(256)
void qkv_proj_kernel(const float* __restrict__ x,
                     const float* __restrict__ w_qk,
                     const float* __restrict__ w_v,
                     unsigned short* __restrict__ qs,
                     unsigned short* __restrict__ kkp,
                     unsigned short* __restrict__ vv) {
  __shared__ __align__(16) __bf16 lds_a[64 * 32];    // [pixel][c]  4 KB
  __shared__ __align__(16) __bf16 lds_w[384 * 32];   // [n][c]     24 KB

  const int tid  = threadIdx.x;
  const int lane = tid & 31, wv = tid >> 5;
  const int lh = lane & 15, hh = lane >> 4;
  const int mblk = blockIdx.x >> 1, nblk = blockIdx.x & 1;
  const int m0 = mblk * 64;          // 64 pixels, one image row segment
  const int n0 = nblk * 384;
  const int b  = m0 >> 14;
  const int h  = (m0 >> 7) & 127;
  const int w0 = m0 & 127;           // 0 or 64

  // per-thread A staging base: channel (tid>>6), pixel column (tid&63)
  const int wi = tid & 63;
  const int cb = tid >> 6;           // 0..3
  const float* xp = x + (((b * Cn + cb) * Hn + h) << 7) + w0 + wi;

  // weight source pointer + compile-time stride (wave-uniform selection)
  const float* wp0; bool qk0;
  { int g = n0 + tid;
    if (g < 512) { wp0 = w_qk + g;        qk0 = true;  }
    else         { wp0 = w_v + (g - 512); qk0 = false; } }
  const float* wp1 = nullptr; bool qk1 = false;
  if (tid < 128) {
    int g = n0 + 256 + tid;
    if (g < 512) { wp1 = w_qk + g;        qk1 = true;  }
    else         { wp1 = w_v + (g - 512); qk1 = false; }
  }

  v8f zero = {};
  v8f acc[4][3];
#pragma unroll
  for (int i = 0; i < 4; ++i)
#pragma unroll
    for (int j = 0; j < 3; ++j) acc[i][j] = zero;

  for (int kt = 0; kt < 8; ++kt) {
    // stage A: 32 channels x 64 pixels; offsets are compile-time immediates
#pragma unroll
    for (int r = 0; r < 8; ++r)
      lds_a[wi * 32 + cb + r * 4] = (__bf16)xp[r * 4 * HWn];
    xp += 32 * HWn;
    // stage W (transposed to [n][c]); strides constant-folded per wave
    if (qk0) stage_wcol<512>(wp0, lds_w + tid * 32);
    else     stage_wcol<256>(wp0, lds_w + tid * 32);
    wp0 += 32 * (qk0 ? 512 : 256);
    if (tid < 128) {
      if (qk1) stage_wcol<512>(wp1, lds_w + (256 + tid) * 32);
      else     stage_wcol<256>(wp1, lds_w + (256 + tid) * 32);
      wp1 += 32 * (qk1 ? 512 : 256);
    }
    __syncthreads();

    v16bf af[4];
#pragma unroll
    for (int mt = 0; mt < 4; ++mt)
      af[mt] = frag_ilv(lds_a + (mt * 16 + lh) * 32, hh);
#pragma unroll
    for (int j = 0; j < 3; ++j) {
      int nt = wv * 3 + j;
      v16bf bf = frag_ctg(lds_w + (nt * 16 + lh) * 32 + hh * 16);
#pragma unroll
      for (int mt = 0; mt < 4; ++mt)
        acc[mt][j] = wmma_bf16(af[mt], bf, acc[mt][j]);
    }
    __syncthreads();
  }

  // scatter C fragments
#pragma unroll
  for (int mt = 0; mt < 4; ++mt)
#pragma unroll
    for (int j = 0; j < 3; ++j) {
      int n = n0 + (wv * 3 + j) * 16 + lh;
#pragma unroll
      for (int i = 0; i < 8; ++i) {
        int ml  = mt * 16 + i + 8 * hh;
        int pix = m0 + ml;
        float v = acc[mt][j][i];
        if (n < 256) {
          qs[pix * 256 + n] = bfbits(v * SCALEF);
        } else if (n < 512) {
          kkp[pix * 256 + (n - 256)] = bfbits(v);
        } else {
          int bb = pix >> 14, hr = (pix >> 7) & 127, ww = pix & 127;
          vv[(((bb * Cn + (n - 512)) * Hn + hr) << 7) + ww] = bfbits(v);
        }
      }
    }
}

// ---------------------------------------------------------------------------
// Kernel 2: soft-superpixel neighborhood attention.
// One block (4 waves, 128 thr) per (8x8 query tile, head). The union of the
// clamped 9x9 windows of an 8x8 tile fits in a 16x16 key tile -> dense GEMM
// + per-element window mask.  p = exp(s)*pi / sum(exp(s)*pi).
// ---------------------------------------------------------------------------
__global__ __launch_bounds__(128)
void attn_kernel(const unsigned short* __restrict__ qs,
                 const unsigned short* __restrict__ kkp,
                 const unsigned short* __restrict__ vv,
                 const float* __restrict__ sims,
                 unsigned short* __restrict__ ao) {
  __shared__ __align__(16) unsigned char smem[49152];
  __bf16* lds_q  = (__bf16*)smem;             // [0,4K)   phase 1
  __bf16* lds_k  = (__bf16*)(smem + 4096);    // [4K,20K) phase 1
  __bf16* lds_p  = (__bf16*)smem;             // [0,32K)  phase 3 (reuses q,k)
  __bf16* lds_vt = (__bf16*)(smem + 32768);   // [32K,48K) V transposed [d][key]

  const int tid  = threadIdx.x;
  const int lane = tid & 31, wv = tid >> 5;
  const int lh = lane & 15, hh = lane >> 4;
  const int head = blockIdx.x & 7;
  const int tile = blockIdx.x >> 3;
  const int b  = tile >> 8;
  const int ty = (tile >> 4) & 15, tx = tile & 15;
  const int h0 = ty * 8, w0 = tx * 8;
  int rbase = h0 - 4; if (rbase < 0) rbase = 0; if (rbase > Hn - 16) rbase = Hn - 16;
  int cbase = w0 - 4; if (cbase < 0) cbase = 0; if (cbase > Wn - 16) cbase = Wn - 16;

  // ---- cooperative loads (async global->LDS when available) ---------------
#ifdef ASYNC_LDS
#pragma unroll
  for (int r = 0; r < 4; ++r) {        // Q: 512 x 16B
    int idx = tid + r * 128;
    int qi = idx >> 2, ch = idx & 3;
    int qh = h0 + (qi >> 3), qw = w0 + (qi & 7);
    async_cp128(qs + ((((b * Hn + qh) << 7) + qw) << 8) + head * HDn + ch * 8,
                (unsigned char*)lds_q + idx * 16);
  }
#pragma unroll
  for (int r = 0; r < 16; ++r) {       // K: 2048 x 16B
    int idx = tid + r * 128;
    int ki = idx >> 2, ch = idx & 3;
    int kr = ki >> 4, kc = ki & 15;
    async_cp128(kkp + ((((b * Hn + rbase + kr) << 7) + (cbase + kc)) << 8) + head * HDn + ch * 8,
                (unsigned char*)lds_k + idx * 16);
  }
#pragma unroll
  for (int r = 0; r < 16; ++r) {       // Vt: 2048 x 8B (rows only 8B aligned)
    int idx = tid + r * 128;
    int row = idx >> 2, qtr = idx & 3;
    int d = row >> 4, kr = row & 15;
    async_cp64(vv + (((b * Cn + head * HDn + d) * Hn + rbase + kr) << 7) + cbase + qtr * 4,
               (unsigned char*)lds_vt + idx * 8);
  }
  async_wait0();
#else
  {   // Q: 128 queries (16 rows x 8 cols) x 32 dims bf16 = 512 x f4
    f4* dst = (f4*)lds_q;
#pragma unroll
    for (int r = 0; r < 4; ++r) {
      int idx = tid + r * 128;
      int qi = idx >> 2, ch = idx & 3;
      int qh = h0 + (qi >> 3), qw = w0 + (qi & 7);
      const f4* src = (const f4*)(qs + ((((b * Hn + qh) << 7) + qw) << 8) + head * HDn);
      dst[idx] = src[ch];
    }
  }
  {   // K: 512 keys (32 rows x 16 cols) x 32 dims = 2048 x f4
    f4* dst = (f4*)lds_k;
#pragma unroll
    for (int r = 0; r < 16; ++r) {
      int idx = tid + r * 128;
      int ki = idx >> 2, ch = idx & 3;
      int kr = ki >> 4, kc = ki & 15;
      const f4* src = (const f4*)(kkp + ((((b * Hn + rbase + kr) << 7) + (cbase + kc)) << 8) + head * HDn);
      dst[idx] = src[ch];
    }
  }
  {   // Vt: [d][key], from V in BCHW (rows only 8B aligned -> f2 copies)
    f2* dst = (f2*)lds_vt;
#pragma unroll
    for (int r = 0; r < 16; ++r) {
      int idx = tid + r * 128;
      int row = idx >> 2, qtr = idx & 3;
      int d = row >> 4, kr = row & 15;
      const f2* src = (const f2*)(vv + (((b * Cn + head * HDn + d) * Hn + rbase + kr) << 7) + cbase);
      dst[idx] = src[qtr];
    }
  }
#endif
  __syncthreads();

  // ---- phase 1: S = Q Kt, 16 WMMAs per wave ------------------------------
  const int mt = wv;  // wave's M tile (16 queries)
  v16bf aq = frag_ilv(lds_q + (mt * 16 + lh) * 32, hh);
  v8f s[16];
#pragma unroll
  for (int nt = 0; nt < 16; ++nt) {
    v16bf bk = frag_ctg(lds_k + (nt * 16 + lh) * 32 + hh * 16);
    v8f z = {};
    s[nt] = wmma_bf16(aq, bk, z);
  }

  // ---- phase 2: masked exp * pi, row-sum, normalize ----------------------
  int ilo[8], jlo[8], sbase[8];
#pragma unroll
  for (int i = 0; i < 8; ++i) {
    int m  = mt * 16 + i + 8 * hh;
    int qh = h0 + (m >> 3), qw = w0 + (m & 7);
    int a = qh - 4; if (a < 0) a = 0; if (a > Hn - KSn) a = Hn - KSn;
    int c = qw - 4; if (c < 0) c = 0; if (c > Wn - KSn) c = Wn - KSn;
    ilo[i] = a; jlo[i] = c;
    sbase[i] = (((b * Hn + qh) << 7) + qw) << 8;
  }
  float rs[8];
#pragma unroll
  for (int i = 0; i < 8; ++i) rs[i] = 0.f;
#pragma unroll
  for (int nt = 0; nt < 16; ++nt) {
    int krg = rbase + nt;          // key row (n>>4 == nt)
    int kcg = cbase + lh;          // key col
    int sp  = ((krg >> 3) << 4) + (kcg >> 3);   // superpixel index
#pragma unroll
    for (int i = 0; i < 8; ++i) {
      bool valid = (krg >= ilo[i]) & (krg < ilo[i] + KSn) &
                   (kcg >= jlo[i]) & (kcg < jlo[i] + KSn);
      float wgt = 0.f;
      if (valid) wgt = __expf(s[nt][i]) * sims[sbase[i] + sp];
      s[nt][i] = wgt;
      rs[i] += wgt;
    }
  }
#pragma unroll
  for (int i = 0; i < 8; ++i) {   // reduce across the 16 lanes of each half
    float v = rs[i];
    v += __shfl_xor(v, 1);
    v += __shfl_xor(v, 2);
    v += __shfl_xor(v, 4);
    v += __shfl_xor(v, 8);
    rs[i] = 1.0f / (v + 1e-12f);
  }
  __syncthreads();                 // everyone done with lds_q / lds_k
#pragma unroll
  for (int nt = 0; nt < 16; ++nt)
#pragma unroll
    for (int i = 0; i < 8; ++i) {
      int m = mt * 16 + i + 8 * hh;
      lds_p[m * 256 + nt * 16 + lh] = (__bf16)(s[nt][i] * rs[i]);
    }
  __syncthreads();

  // ---- phase 3: O = P V, 16 WMMAs per wave -------------------------------
  v8f o0 = {}, o1 = {};
  const __bf16* prow = lds_p + (mt * 16 + lh) * 256;
#pragma unroll
  for (int ks = 0; ks < 8; ++ks) {
    v16bf ap = frag_ilv(prow + ks * 32, hh);
    v16bf b0 = frag_ctg(lds_vt + (0 * 16 + lh) * 256 + ks * 32 + hh * 16);
    v16bf b1 = frag_ctg(lds_vt + (1 * 16 + lh) * 256 + ks * 32 + hh * 16);
    o0 = wmma_bf16(ap, b0, o0);
    o1 = wmma_bf16(ap, b1, o1);
  }
#pragma unroll
  for (int i = 0; i < 8; ++i) {
    int m  = mt * 16 + i + 8 * hh;
    int qh = h0 + (m >> 3), qw = w0 + (m & 7);
    int base = ((((b * Hn + qh) << 7) + qw) << 8) + head * HDn;
    ao[base + lh]      = bfbits(o0[i]);
    ao[base + 16 + lh] = bfbits(o1[i]);
  }
}

// ---------------------------------------------------------------------------
// Kernel 3: output projection [65536 x 256] @ w_proj(256x256) -> f32 BCHW.
// ---------------------------------------------------------------------------
__global__ __launch_bounds__(256)
void out_proj_kernel(const unsigned short* __restrict__ ao,
                     const float* __restrict__ w_proj,
                     float* __restrict__ out) {
  __shared__ __align__(16) __bf16 lds_a[64 * 32];    //  4 KB
  __shared__ __align__(16) __bf16 lds_w[256 * 32];   // 16 KB [n][c]

  const int tid  = threadIdx.x;
  const int lane = tid & 31, wv = tid >> 5;
  const int lh = lane & 15, hh = lane >> 4;
  const int m0 = blockIdx.x * 64;
  const int qi = tid >> 2, ch = tid & 3;
  const float* wp = w_proj + tid;                       // column n = tid
  const unsigned short* ap = ao + (m0 + qi) * 256 + ch * 8;

  v8f zero = {};
  v8f acc[4][2];
#pragma unroll
  for (int i = 0; i < 4; ++i)
#pragma unroll
    for (int j = 0; j < 2; ++j) acc[i][j] = zero;

  for (int kt = 0; kt < 8; ++kt) {
    // A tile (already bf16): 64 x 32 = 256 x 16B, one per thread
#ifdef ASYNC_LDS
    async_cp128(ap, (unsigned char*)lds_a + tid * 16);
#else
    ((f4*)lds_a)[tid] = *(const f4*)ap;
#endif
    ap += 32;
    stage_wcol<256>(wp, lds_w + tid * 32);   // w_proj[c][n] -> lds_w[n][c]
    wp += 32 * 256;
#ifdef ASYNC_LDS
    async_wait0();
#endif
    __syncthreads();

    v16bf af[4];
#pragma unroll
    for (int mtI = 0; mtI < 4; ++mtI)
      af[mtI] = frag_ilv(lds_a + (mtI * 16 + lh) * 32, hh);
#pragma unroll
    for (int j = 0; j < 2; ++j) {
      int nt = wv * 2 + j;
      v16bf bf = frag_ctg(lds_w + (nt * 16 + lh) * 32 + hh * 16);
#pragma unroll
      for (int mtI = 0; mtI < 4; ++mtI)
        acc[mtI][j] = wmma_bf16(af[mtI], bf, acc[mtI][j]);
    }
    __syncthreads();
  }

#pragma unroll
  for (int mtI = 0; mtI < 4; ++mtI)
#pragma unroll
    for (int j = 0; j < 2; ++j) {
      int chn = (wv * 2 + j) * 16 + lh;
#pragma unroll
      for (int i = 0; i < 8; ++i) {
        int pix = m0 + mtI * 16 + i + 8 * hh;
        int bb = pix >> 14, hr = (pix >> 7) & 127, ww = pix & 127;
        out[(((bb * Cn + chn) * Hn + hr) << 7) + ww] = acc[mtI][j][i];
      }
    }
}

// ---------------------------------------------------------------------------
extern "C" void kernel_launch(void* const* d_in, const int* in_sizes, int n_in,
                              void* d_out, int out_size, void* d_ws, size_t ws_size,
                              hipStream_t stream) {
  const float* x      = (const float*)d_in[0];
  const float* sims   = (const float*)d_in[1];
  const float* w_qk   = (const float*)d_in[2];
  const float* w_v    = (const float*)d_in[3];
  const float* w_proj = (const float*)d_in[4];
  float* out = (float*)d_out;

  // workspace: Q, K (BHWC bf16), V (BCHW bf16), attn out (BHWC bf16)
  unsigned short* qs  = (unsigned short*)d_ws;
  unsigned short* kkp = qs  + (size_t)NPIX * Cn;
  unsigned short* vv  = kkp + (size_t)NPIX * Cn;
  unsigned short* ao  = vv  + (size_t)NPIX * Cn;

  qkv_proj_kernel<<<dim3(2048), dim3(256), 0, stream>>>(x, w_qk, w_v, qs, kkp, vv);
  attn_kernel<<<dim3(8192), dim3(128), 0, stream>>>(qs, kkp, vv, sims, ao);
  out_proj_kernel<<<dim3(1024), dim3(256), 0, stream>>>(ao, w_proj, out);
}
